// SigmoidMoE_51436528337283
// MI455X (gfx1250) — matrix operand order repaired and checked by reading, not compiled, not run on hardware
//
#include <hip/hip_runtime.h>
#include <hip/hip_bf16.h>

typedef __attribute__((ext_vector_type(16))) __bf16 v16bf;
typedef __attribute__((ext_vector_type(8)))  float  v8f;
typedef __attribute__((ext_vector_type(4)))  __bf16 v4bf;
typedef __attribute__((ext_vector_type(2)))  __bf16 v2bf;
typedef __attribute__((ext_vector_type(4)))  unsigned tdmv4u;
typedef __attribute__((ext_vector_type(8)))  int      tdmv8i;
typedef __attribute__((ext_vector_type(4)))  int      tdmv4i;

#define TOKENS 4096
#define DIMD   1024
#define HIDD   2048
#define H2D    4096
#define NEXP   8

union FragBF { v16bf v; unsigned u[8]; };

static __device__ __forceinline__ v8f wmma_bf16(v16bf a, v16bf b, v8f c) {
  return __builtin_amdgcn_wmma_f32_16x16x32_bf16(false, a, false, b, (short)0, c,
                                                 false, false);
}

// pack two floats -> packed bf16 dword (a -> [15:0], b -> [31:16])
static __device__ __forceinline__ unsigned pack2bf(float a, float b) {
#if __has_builtin(__builtin_amdgcn_cvt_pk_bf16_f32)
  union { v2bf v; unsigned u; } r;
  r.v = __builtin_amdgcn_cvt_pk_bf16_f32(a, b);
  return r.u;
#else
  union { __bf16 h; unsigned short s; } x, y;
  x.h = (__bf16)a; y.h = (__bf16)b;
  return (unsigned)x.s | ((unsigned)y.s << 16);
#endif
}

// TDM: load a [validRows x 64B] bf16 tile (pitch HIDD*2 bytes) into LDS.
// data_size=4B, tile = 16 dwords x 8 rows, OOB rows (>= validRows) read as 0.
static __device__ __forceinline__ void tdm_load_a(const void* gsrc, void* ldsDst,
                                                  int validRows) {
  unsigned ldsa = (unsigned)(uintptr_t)ldsDst;
  unsigned long long ga = (unsigned long long)(uintptr_t)gsrc;
  unsigned galo = (unsigned)ga;
  unsigned gahi = (unsigned)((ga >> 32) & 0x01FFFFFFull) | 0x80000000u; // type=2
  unsigned td1 = (validRows < 0) ? 0u : (unsigned)validRows;
  ldsa = __builtin_amdgcn_readfirstlane(ldsa);
  galo = __builtin_amdgcn_readfirstlane(galo);
  gahi = __builtin_amdgcn_readfirstlane(gahi);
  td1  = __builtin_amdgcn_readfirstlane(td1);
  tdmv4u g0;
  g0[0] = 1u;        // count=1 valid descriptor
  g0[1] = ldsa;      // lds_addr
  g0[2] = galo;      // global_addr[31:0]
  g0[3] = gahi;      // global_addr[56:32] | type=2<<30
  tdmv8i g1;
  g1[0] = 0x00020000;                        // data_size = 4B, no flags
  g1[1] = (int)(16u << 16);                  // tensor_dim0 = 16 dwords
  g1[2] = (int)((td1 & 0xFFFFu) << 16);      // tensor_dim1 lo
  g1[3] = (int)((td1 >> 16) | (16u << 16));  // tensor_dim1 hi | tile_dim0 = 16
  g1[4] = 8;                                 // tile_dim1 = 8 rows, tile_dim2 = 0
  g1[5] = (int)(HIDD / 2);                   // tensor_dim0_stride = 1024 dwords
  g1[6] = 0;
  g1[7] = 0;
  tdmv4i gz = {0, 0, 0, 0};
#if defined(__clang_major__) && __clang_major__ >= 23
  tdmv8i gz8 = {0, 0, 0, 0, 0, 0, 0, 0};
  __builtin_amdgcn_tensor_load_to_lds(g0, g1, gz, gz, gz8, 0);
#else
  __builtin_amdgcn_tensor_load_to_lds(g0, g1, gz, gz, 0);
#endif
}

// ---------------------------------------------------------------- x -> bf16
__global__ __launch_bounds__(256) void moe_cvt_x(const float* __restrict__ x,
                                                 __bf16* __restrict__ xbf) {
  size_t i = ((size_t)blockIdx.x * 256 + threadIdx.x) * 4;
  float4 f = *(const float4*)(x + i);
  uint2 o;
  o.x = pack2bf(f.x, f.y);
  o.y = pack2bf(f.z, f.w);
  *(uint2*)(xbf + i) = o;
}

// ---------------------------------------------------------------- router
__global__ __launch_bounds__(256) void moe_router(
    const float* __restrict__ x, const float* __restrict__ rw,
    const float* __restrict__ rb, int* __restrict__ tokE,
    float* __restrict__ tokW, float* __restrict__ auxP) {
  const int t = blockIdx.x;
  const int tid = threadIdx.x;
  __shared__ float red[256 * 8];
  float p[8];
#pragma unroll
  for (int e = 0; e < 8; ++e) p[e] = 0.0f;
  const float* xr = x + (size_t)t * DIMD;
  for (int d = tid; d < DIMD; d += 256) {
    float xv = xr[d];
    const float* w = rw + (size_t)d * 8;
#pragma unroll
    for (int e = 0; e < 8; ++e) p[e] += xv * w[e];
  }
#pragma unroll
  for (int e = 0; e < 8; ++e) red[tid * 8 + e] = p[e];
  __syncthreads();
  for (int off = 128; off > 0; off >>= 1) {
    if (tid < off) {
#pragma unroll
      for (int e = 0; e < 8; ++e) red[tid * 8 + e] += red[(tid + off) * 8 + e];
    }
    __syncthreads();
  }
  if (tid == 0) {
    float l[8], s[8], aux = 0.0f;
#pragma unroll
    for (int e = 0; e < 8; ++e) {
      l[e] = red[e] + rb[e];
      aux += l[e] * l[e];
      s[e] = 1.0f / (1.0f + __expf(-l[e]));
    }
    int i0 = 0;
#pragma unroll
    for (int e = 1; e < 8; ++e) if (s[e] > s[i0]) i0 = e;  // first-index tiebreak
    int i1 = (i0 == 0) ? 1 : 0;
#pragma unroll
    for (int e = 0; e < 8; ++e) if (e != i0 && s[e] > s[i1]) i1 = e;
    float v0 = s[i0], v1 = s[i1];
    float dn = v0 + v1 + 1e-6f;
    tokE[2 * t] = i0;      tokE[2 * t + 1] = i1;
    tokW[2 * t] = v0 / dn; tokW[2 * t + 1] = v1 / dn;
    auxP[t] = aux;
  }
}

// ---------------------------------------------------------------- list build
__global__ __launch_bounds__(256) void moe_assign(
    const int* __restrict__ tokE, const float* __restrict__ tokW,
    int* __restrict__ counts, int* __restrict__ offs,
    int* __restrict__ tokenOfRow, int* __restrict__ rowOf,
    float* __restrict__ wOf) {
  __shared__ int lcnt[8], loff[8], lpos[8];
  const int tid = threadIdx.x;
  if (tid < 8) lcnt[tid] = 0;
  __syncthreads();
  for (int t = tid; t < TOKENS; t += 256) {
    atomicAdd(&lcnt[tokE[2 * t]], 1);
    atomicAdd(&lcnt[tokE[2 * t + 1]], 1);
  }
  __syncthreads();
  if (tid == 0) {
    int acc = 0;
    for (int e = 0; e < 8; ++e) { loff[e] = acc; acc += lcnt[e]; }
  }
  __syncthreads();
  if (tid < 8) { lpos[tid] = 0; counts[tid] = lcnt[tid]; offs[tid] = loff[tid]; }
  __syncthreads();
  for (int t = tid; t < TOKENS; t += 256) {
#pragma unroll
    for (int k = 0; k < 2; ++k) {
      int e = tokE[2 * t + k];
      int pos = atomicAdd(&lpos[e], 1);
      int row = loff[e] + pos;
      tokenOfRow[row] = t;
      rowOf[2 * t + k] = row;
      wOf[2 * t + k] = tokW[2 * t + k];
    }
  }
}

// ---------------------------------------------------------------- GEMM1 + SwiGLU
__global__ __launch_bounds__(256) void moe_gemm1(
    const __bf16* __restrict__ xbf, const float* __restrict__ w12,
    const int* __restrict__ counts, const int* __restrict__ offs,
    const int* __restrict__ tokenOfRow, __bf16* __restrict__ gbuf) {
  const int e = blockIdx.z;
  const int cnt = counts[e];
  const int row0 = blockIdx.y * 64;
  if (row0 >= cnt) return;
  const int n0 = blockIdx.x * 64;
  const int base = offs[e];

  __shared__ __align__(16) __bf16 sA[2][64 * 32];   // [m][k]
  __shared__ __align__(16) __bf16 sB1[2][64 * 32];  // [n][k] transposed
  __shared__ __align__(16) __bf16 sB2[2][64 * 32];
  __shared__ int sTok[64];

  const int tid = threadIdx.x;
  if (tid < 64) {
    int r = row0 + tid;
    sTok[tid] = tokenOfRow[base + ((r < cnt) ? r : 0)];
  }
  __syncthreads();

  const int lane = tid & 31;
  const int wave = tid >> 5;
  const int rT = wave >> 1;
  const int cp = (wave & 1) * 2;

  v8f vz = {0.f, 0.f, 0.f, 0.f, 0.f, 0.f, 0.f, 0.f};
  v8f acc1[2], acc2[2];
  acc1[0] = vz; acc1[1] = vz; acc2[0] = vz; acc2[1] = vz;

  const int aRow = tid >> 2;
  const int aSeg = (tid & 3) * 8;
  const int n4 = (tid & 15) * 4;          // 4 output columns
  const int k2 = ((tid >> 4) & 15) * 2;   // K pair
  const float* B1base = w12 + (size_t)e * DIMD * H2D + n0;
  const float* B2base = B1base + HIDD;
  const __bf16* aGlob = xbf + (size_t)sTok[aRow] * DIMD + aSeg;

  const int m = lane & 15;
  const int hf = lane >> 4;
  const int nl = lane & 15;
  const int hk16 = lane & 16;

  // ---- prologue: tile k0 = 0 into buffer 0
  uint4 aR = *(const uint4*)(aGlob);
  const float* p1 = B1base + (size_t)k2 * H2D + n4;
  const float* p2 = B2base + (size_t)k2 * H2D + n4;
  float4 b1r0 = *(const float4*)p1, b1r1 = *(const float4*)(p1 + H2D);
  float4 b2r0 = *(const float4*)p2, b2r1 = *(const float4*)(p2 + H2D);
  *(uint4*)&sA[0][aRow * 32 + aSeg] = aR;
  *(unsigned*)&sB1[0][(n4 + 0) * 32 + k2] = pack2bf(b1r0.x, b1r1.x);
  *(unsigned*)&sB1[0][(n4 + 1) * 32 + k2] = pack2bf(b1r0.y, b1r1.y);
  *(unsigned*)&sB1[0][(n4 + 2) * 32 + k2] = pack2bf(b1r0.z, b1r1.z);
  *(unsigned*)&sB1[0][(n4 + 3) * 32 + k2] = pack2bf(b1r0.w, b1r1.w);
  *(unsigned*)&sB2[0][(n4 + 0) * 32 + k2] = pack2bf(b2r0.x, b2r1.x);
  *(unsigned*)&sB2[0][(n4 + 1) * 32 + k2] = pack2bf(b2r0.y, b2r1.y);
  *(unsigned*)&sB2[0][(n4 + 2) * 32 + k2] = pack2bf(b2r0.z, b2r1.z);
  *(unsigned*)&sB2[0][(n4 + 3) * 32 + k2] = pack2bf(b2r0.w, b2r1.w);
  __syncthreads();

  int cur = 0;
  for (int k0 = 0; k0 < DIMD; k0 += 32) {
    const bool hasNext = (k0 + 32) < DIMD;
    if (hasNext) {  // issue next tile's global loads early (latency hiding)
      int kn = k0 + 32;
      aR = *(const uint4*)(aGlob + kn);
      p1 = B1base + (size_t)(kn + k2) * H2D + n4;
      p2 = B2base + (size_t)(kn + k2) * H2D + n4;
      b1r0 = *(const float4*)p1; b1r1 = *(const float4*)(p1 + H2D);
      b2r0 = *(const float4*)p2; b2r1 = *(const float4*)(p2 + H2D);
      if (kn + 32 < DIMD) {  // speculative prefetch of the tile after next
        __builtin_prefetch((const void*)(p1 + 32 * H2D), 0, 0);
        __builtin_prefetch((const void*)(p2 + 32 * H2D), 0, 0);
      }
    }

    const __bf16* As = sA[cur];
    const __bf16* B1s = sB1[cur];
    const __bf16* B2s = sB2[cur];
    FragBF a;
#pragma unroll
    for (int j = 0; j < 8; ++j) {
      int kb = (hf << 4) | (j << 1);
      a.u[j] = *(const unsigned*)&As[(rT * 16 + m) * 32 + kb];
    }
    FragBF b10, b11, b20, b21;
    const int nr0 = ((cp + 0) * 16 + nl) * 32;
    const int nr1 = ((cp + 1) * 16 + nl) * 32;
#pragma unroll
    for (int j = 0; j < 8; ++j) {
      int kp = hk16 | (j << 1);
      b10.u[j] = *(const unsigned*)&B1s[nr0 + kp];
      b11.u[j] = *(const unsigned*)&B1s[nr1 + kp];
      b20.u[j] = *(const unsigned*)&B2s[nr0 + kp];
      b21.u[j] = *(const unsigned*)&B2s[nr1 + kp];
    }
    acc1[0] = wmma_bf16(a.v, b10.v, acc1[0]);
    acc1[1] = wmma_bf16(a.v, b11.v, acc1[1]);
    acc2[0] = wmma_bf16(a.v, b20.v, acc2[0]);
    acc2[1] = wmma_bf16(a.v, b21.v, acc2[1]);

    if (hasNext) {
      const int nxt = cur ^ 1;
      *(uint4*)&sA[nxt][aRow * 32 + aSeg] = aR;
      *(unsigned*)&sB1[nxt][(n4 + 0) * 32 + k2] = pack2bf(b1r0.x, b1r1.x);
      *(unsigned*)&sB1[nxt][(n4 + 1) * 32 + k2] = pack2bf(b1r0.y, b1r1.y);
      *(unsigned*)&sB1[nxt][(n4 + 2) * 32 + k2] = pack2bf(b1r0.z, b1r1.z);
      *(unsigned*)&sB1[nxt][(n4 + 3) * 32 + k2] = pack2bf(b1r0.w, b1r1.w);
      *(unsigned*)&sB2[nxt][(n4 + 0) * 32 + k2] = pack2bf(b2r0.x, b2r1.x);
      *(unsigned*)&sB2[nxt][(n4 + 1) * 32 + k2] = pack2bf(b2r0.y, b2r1.y);
      *(unsigned*)&sB2[nxt][(n4 + 2) * 32 + k2] = pack2bf(b2r0.z, b2r1.z);
      *(unsigned*)&sB2[nxt][(n4 + 3) * 32 + k2] = pack2bf(b2r0.w, b2r1.w);
    }
    __syncthreads();
    cur ^= 1;
  }

  // epilogue: g = silu(x1) * x2 -> bf16 compact buffer [8192, 2048]
#pragma unroll
  for (int c = 0; c < 2; ++c) {
#pragma unroll
    for (int j = 0; j < 8; ++j) {
      int rowl = rT * 16 + j + (hf << 3);
      if (row0 + rowl < cnt) {
        float x1 = acc1[c][j];
        float x2 = acc2[c][j];
        float gg = (x1 / (1.0f + __expf(-x1))) * x2;
        int col = n0 + (cp + c) * 16 + nl;
        gbuf[(size_t)(base + row0 + rowl) * HIDD + col] = (__bf16)gg;
      }
    }
  }
}

// ---------------------------------------------------------------- GEMM2 (TDM A)
__global__ __launch_bounds__(256) void moe_gemm2(
    const __bf16* __restrict__ gbuf, const float* __restrict__ w3,
    const int* __restrict__ counts, const int* __restrict__ offs,
    float* __restrict__ eout) {
  const int e = blockIdx.z;
  const int cnt = counts[e];
  const int row0 = blockIdx.y * 64;
  if (row0 >= cnt) return;
  const int n0 = blockIdx.x * 64;
  const int base = offs[e];

  __shared__ __align__(16) __bf16 sA[2][64 * 32];
  __shared__ __align__(16) __bf16 sB[2][64 * 32];  // [n][k]

  const int tid = threadIdx.x;
  const int lane = tid & 31;
  const int wave = tid >> 5;
  const int rT = wave >> 1;
  const int cp = (wave & 1) * 2;

  v8f vz = {0.f, 0.f, 0.f, 0.f, 0.f, 0.f, 0.f, 0.f};
  v8f acc[2];
  acc[0] = vz; acc[1] = vz;

  const int n4 = (tid & 15) * 4;
  const int k2 = ((tid >> 4) & 15) * 2;
  const float* Bbase = w3 + (size_t)e * HIDD * DIMD + n0;
  // each wave TDM-loads its own 8-row sub-tile of A
  const int waveRow = row0 + wave * 8;
  const __bf16* aWaveGlob = gbuf + (size_t)(base + waveRow) * HIDD;
  const int waveValid = cnt - waveRow;  // rows >= this come back as zeros

  const int m = lane & 15;
  const int hf = lane >> 4;
  const int nl = lane & 15;
  const int hk16 = lane & 16;

  // ---- prologue: tile k0 = 0 into buffer 0
  tdm_load_a(aWaveGlob, &sA[0][wave * 8 * 32], waveValid);
  const float* bp = Bbase + (size_t)k2 * DIMD + n4;
  float4 br0 = *(const float4*)bp, br1 = *(const float4*)(bp + DIMD);
  *(unsigned*)&sB[0][(n4 + 0) * 32 + k2] = pack2bf(br0.x, br1.x);
  *(unsigned*)&sB[0][(n4 + 1) * 32 + k2] = pack2bf(br0.y, br1.y);
  *(unsigned*)&sB[0][(n4 + 2) * 32 + k2] = pack2bf(br0.z, br1.z);
  *(unsigned*)&sB[0][(n4 + 3) * 32 + k2] = pack2bf(br0.w, br1.w);
  __builtin_amdgcn_s_wait_tensorcnt(0);
  __syncthreads();

  int cur = 0;
  for (int k0 = 0; k0 < HIDD; k0 += 32) {
    const bool hasNext = (k0 + 32) < HIDD;
    if (hasNext) {
      const int kn = k0 + 32;
      tdm_load_a(aWaveGlob + kn, &sA[cur ^ 1][wave * 8 * 32], waveValid);
      bp = Bbase + (size_t)(kn + k2) * DIMD + n4;
      br0 = *(const float4*)bp; br1 = *(const float4*)(bp + DIMD);
      if (kn + 32 < HIDD) {
        __builtin_prefetch((const void*)(bp + 32 * DIMD), 0, 0);
      }
    }

    const __bf16* As = sA[cur];
    const __bf16* Bs = sB[cur];
    FragBF a;
#pragma unroll
    for (int j = 0; j < 8; ++j) {
      int kb = (hf << 4) | (j << 1);
      a.u[j] = *(const unsigned*)&As[(rT * 16 + m) * 32 + kb];
    }
    FragBF b0, b1;
    const int nr0 = ((cp + 0) * 16 + nl) * 32;
    const int nr1 = ((cp + 1) * 16 + nl) * 32;
#pragma unroll
    for (int j = 0; j < 8; ++j) {
      int kp = hk16 | (j << 1);
      b0.u[j] = *(const unsigned*)&Bs[nr0 + kp];
      b1.u[j] = *(const unsigned*)&Bs[nr1 + kp];
    }
    acc[0] = wmma_bf16(a.v, b0.v, acc[0]);
    acc[1] = wmma_bf16(a.v, b1.v, acc[1]);

    if (hasNext) {
      const int nxt = cur ^ 1;
      *(unsigned*)&sB[nxt][(n4 + 0) * 32 + k2] = pack2bf(br0.x, br1.x);
      *(unsigned*)&sB[nxt][(n4 + 1) * 32 + k2] = pack2bf(br0.y, br1.y);
      *(unsigned*)&sB[nxt][(n4 + 2) * 32 + k2] = pack2bf(br0.z, br1.z);
      *(unsigned*)&sB[nxt][(n4 + 3) * 32 + k2] = pack2bf(br0.w, br1.w);
    }
    __builtin_amdgcn_s_wait_tensorcnt(0);
    __syncthreads();
    cur ^= 1;
  }

#pragma unroll
  for (int c = 0; c < 2; ++c) {
#pragma unroll
    for (int j = 0; j < 8; ++j) {
      int rowl = rT * 16 + j + (hf << 3);
      if (row0 + rowl < cnt) {
        eout[(size_t)(base + row0 + rowl) * DIMD + n0 + (cp + c) * 16 + nl] =
            acc[c][j];
      }
    }
  }
}

// ---------------------------------------------------------------- combine
__global__ __launch_bounds__(256) void moe_combine(
    const float* __restrict__ eout, const int* __restrict__ rowOf,
    const float* __restrict__ wOf, float* __restrict__ out) {
  const int t = blockIdx.x;
  const int tid = threadIdx.x;
  const int r0 = rowOf[2 * t], r1 = rowOf[2 * t + 1];
  const float w0 = wOf[2 * t], w1 = wOf[2 * t + 1];
  const float* e0 = eout + (size_t)r0 * DIMD;
  const float* e1 = eout + (size_t)r1 * DIMD;
  float* o = out + (size_t)t * DIMD;
  for (int d = tid; d < DIMD; d += 256) o[d] = w0 * e0[d] + w1 * e1[d];
}

// ---------------------------------------------------------------- aux loss
__global__ __launch_bounds__(256) void moe_aux(const float* __restrict__ auxP,
                                               float* __restrict__ outAux) {
  __shared__ float s[256];
  const int tid = threadIdx.x;
  float a = 0.0f;
  for (int i = tid; i < TOKENS; i += 256) a += auxP[i];
  s[tid] = a;
  __syncthreads();
  for (int off = 128; off > 0; off >>= 1) {
    if (tid < off) s[tid] += s[tid + off];
    __syncthreads();
  }
  if (tid == 0) *outAux = 0.01f * s[0] / (float)(TOKENS * NEXP);
}

// ---------------------------------------------------------------- launch
extern "C" void kernel_launch(void* const* d_in, const int* in_sizes, int n_in,
                              void* d_out, int out_size, void* d_ws,
                              size_t ws_size, hipStream_t stream) {
  const float* x   = (const float*)d_in[0];
  const float* rw  = (const float*)d_in[1];
  const float* rb  = (const float*)d_in[2];
  const float* w12 = (const float*)d_in[3];
  const float* w3  = (const float*)d_in[4];
  float* out = (float*)d_out;

  char* ws = (char*)d_ws;
  __bf16* xbf      = (__bf16*)(ws + 0);           // 8,388,608 B
  __bf16* gbuf     = (__bf16*)(ws + 8388608);     // 33,554,432 B
  float*  eout     = (float*) (ws + 41943040);    // 33,554,432 B
  int*    tokE     = (int*)   (ws + 75497472);    // 32,768 B
  float*  tokW     = (float*) (ws + 75530240);    // 32,768 B
  float*  auxP     = (float*) (ws + 75563008);    // 16,384 B
  int*    counts   = (int*)   (ws + 75579392);    // 64 B
  int*    offs     = (int*)   (ws + 75579456);    // 64 B
  int*    tokenOfRow = (int*) (ws + 75579520);    // 32,768 B
  int*    rowOf    = (int*)   (ws + 75612288);    // 32,768 B
  float*  wOf      = (float*) (ws + 75645056);    // 32,768 B  (total ~75.7 MB)

  moe_cvt_x<<<4096, 256, 0, stream>>>(x, xbf);
  moe_router<<<TOKENS, 256, 0, stream>>>(x, rw, rb, tokE, tokW, auxP);
  moe_assign<<<1, 256, 0, stream>>>(tokE, tokW, counts, offs, tokenOfRow, rowOf,
                                    wOf);
  moe_gemm1<<<dim3(H2D / 2 / 64, 64, NEXP), 256, 0, stream>>>(
      xbf, w12, counts, offs, tokenOfRow, gbuf);
  moe_gemm2<<<dim3(DIMD / 64, 64, NEXP), 256, 0, stream>>>(gbuf, w3, counts,
                                                           offs, eout);
  moe_combine<<<TOKENS, 256, 0, stream>>>(eout, rowOf, wOf, out);
  moe_aux<<<1, 256, 0, stream>>>(auxP, out + (size_t)TOKENS * DIMD);
}
